// DecoderRNN_13520557048047
// MI455X (gfx1250) — matrix operand order repaired and checked by reading, not compiled, not run on hardware
//
#include <hip/hip_runtime.h>
#include <stddef.h>
#include <stdint.h>

// ---------------- problem constants ----------------
#define E 512
#define H 1024
#define V 50257
#define NUM_STEPS 19
#define VPAD 50272              // V rounded up to multiple of 16
#define NTILES (VPAD / 16)      // 3142 WMMA row-tiles
#define NEG_BIG (-3.0e38f)      // argmax floor / pad-row bias

// ---------------- vector types for WMMA ----------------
typedef __attribute__((ext_vector_type(16))) __bf16    v16bf;
typedef __attribute__((ext_vector_type(8)))  float     v8f;
typedef __attribute__((ext_vector_type(4)))  unsigned  u32x4;

union ABv { v16bf v; u32x4 q[2]; };   // 32 bytes = 8 VGPRs per lane

// ---------------- helpers ----------------
__device__ __forceinline__ unsigned short f2bf(float f) {
    // round-to-nearest-even fp32 -> bf16
    unsigned u = __float_as_uint(f);
    unsigned r = u + 0x7FFFu + ((u >> 16) & 1u);
    return (unsigned short)(r >> 16);
}
__device__ __forceinline__ unsigned fkey(float f) {
    // monotonic float -> uint mapping (finite values)
    unsigned u = __float_as_uint(f);
    return (u & 0x80000000u) ? ~u : (u | 0x80000000u);
}
__device__ __forceinline__ float sigmoidf(float x) {
    return 1.0f / (1.0f + __expf(-x));
}

// ---------------- kernels ----------------

// One-time (per launch) fp32 -> bf16 conversion of W_lin, zero-padding rows V..VPAD-1.
__global__ void convert_wlin_kernel(const float* __restrict__ W_lin,
                                    unsigned short* __restrict__ wbf) {
    const size_t e = ((size_t)blockIdx.x * blockDim.x + threadIdx.x) * 4;
    const size_t VE = (size_t)V * H;
    if (e < VE) {
        const float4 f = *(const float4*)(W_lin + e);
        unsigned short* d = wbf + e;
        d[0] = f2bf(f.x); d[1] = f2bf(f.y); d[2] = f2bf(f.z); d[3] = f2bf(f.w);
    } else {
        unsigned short* d = wbf + e;   // pad rows -> zero
        d[0] = 0; d[1] = 0; d[2] = 0; d[3] = 0;
    }
}

// Padded bias: bpad[i] = b_lin[i] for i < V, else -inf-ish (pad rows lose argmax).
__global__ void bias_pad_kernel(const float* __restrict__ b_lin,
                                float* __restrict__ bpad) {
    const int i = blockIdx.x * blockDim.x + threadIdx.x;
    if (i < VPAD) bpad[i] = (i < V) ? b_lin[i] : NEG_BIG;
}

// x = inp, h = 0, c = 0
__global__ void init_state_kernel(const float* __restrict__ inp,
                                  float* __restrict__ x,
                                  float* __restrict__ h,
                                  float* __restrict__ c) {
    const int tid = blockIdx.x * blockDim.x + threadIdx.x;
    if (tid < E) x[tid] = inp[tid];
    if (tid < H) { h[tid] = 0.0f; c[tid] = 0.0f; }
}

// gates[j] = dot(W_ih[j,:], x) + dot(W_hh[j,:], h) + b_ih[j] + b_hh[j]
// one wave per output row; fp32 for recurrent numerical safety (24MB, L2 resident)
__global__ void gates_kernel(const float* __restrict__ W_ih,
                             const float* __restrict__ W_hh,
                             const float* __restrict__ b_ih,
                             const float* __restrict__ b_hh,
                             const float* __restrict__ x,
                             const float* __restrict__ h,
                             float* __restrict__ gates) {
    const int row  = blockIdx.x * (blockDim.x >> 5) + (threadIdx.x >> 5);
    const int lane = threadIdx.x & 31;
    const float* wi = W_ih + (size_t)row * E;
    const float* wh = W_hh + (size_t)row * H;
    float acc = 0.0f;
    #pragma unroll 4
    for (int k = lane; k < E; k += 32) acc = fmaf(wi[k], x[k], acc);
    #pragma unroll 4
    for (int k = lane; k < H; k += 32) acc = fmaf(wh[k], h[k], acc);
    #pragma unroll
    for (int off = 16; off; off >>= 1) acc += __shfl_down(acc, off, 32);
    if (lane == 0) gates[row] = acc + b_ih[row] + b_hh[row];
}

// LSTM cell pointwise update; also produce bf16 h and reset the argmax cell
__global__ void cell_kernel(const float* __restrict__ gates,
                            float* __restrict__ c,
                            float* __restrict__ h,
                            unsigned short* __restrict__ hbf,
                            unsigned long long* __restrict__ key) {
    const int tid = blockIdx.x * blockDim.x + threadIdx.x;
    if (tid < H) {
        const float ig = sigmoidf(gates[tid]);
        const float fg = sigmoidf(gates[H + tid]);
        const float gg = tanhf(gates[2 * H + tid]);
        const float og = sigmoidf(gates[3 * H + tid]);
        const float cn = fmaf(fg, c[tid], ig * gg);
        c[tid] = cn;
        const float hn = og * tanhf(cn);
        h[tid]  = hn;
        hbf[tid] = f2bf(hn);
    }
    if (tid == 0) *key = 0ull;   // floor for packed argmax
}

// logits GEMV via v_wmma_f32_16x16x32_bf16 + fused argmax.
// A = 16 rows of W_lin (bf16), B = h chunk broadcast into all 16 columns.
// C is *initialized with the bias* (pad rows = -inf), so after the K loop each
// lane holds 8 finished logits -> branchless register argmax + wave reduction.
__global__ void logits_argmax_kernel(const unsigned short* __restrict__ wbf,
                                     const unsigned short* __restrict__ hbf,
                                     const float* __restrict__ bpad,
                                     unsigned long long* __restrict__ key) {
    const int wave = (blockIdx.x * blockDim.x + threadIdx.x) >> 5;
    const int lane = threadIdx.x & 31;
    if (wave >= NTILES) return;

    const int vbase = wave * 16;
    const int hi    = lane >> 4;                       // lane half: 0 or 1
    const int row   = vbase + (lane & 15);             // A-matrix M = lane (mod 16)
    const int mbase = vbase + hi * 8;                  // first logit owned by this lane
    // A 16-bit layout: lanes 0-15 hold K {0..7,16..23}; lanes 16-31 hold K {8..15,24..31}
    const unsigned short* arow = wbf + (size_t)row * H + (hi << 3);
    // B 16-bit layout: lanes 0-15 hold K {0..7,16..23}; lanes 16-31 hold K {8..15,24..31} (all N equal = h)
    const unsigned short* brow = hbf + (hi << 3);

    // C/D layout: VGPR r, lanes 0-15 -> M=r ; lanes 16-31 -> M=8+r.
    // Seed accumulator with bias for rows mbase..mbase+7 (32B aligned).
    v8f acc = *(const v8f*)(bpad + mbase);

    #pragma unroll 4
    for (int kb = 0; kb < H; kb += 32) {
        ABv a, b;
        a.q[0] = *(const u32x4*)(arow + kb);           // K = kb + {0..7} (+8 if hi)
        a.q[1] = *(const u32x4*)(arow + kb + 16);      // K = kb + {16..23} (+8 if hi)
        b.q[0] = *(const u32x4*)(brow + kb);           // K = kb + {0..7} (+8 if hi)
        b.q[1] = *(const u32x4*)(brow + kb + 16);      // K = kb + {16..23} (+8 if hi)
        acc = __builtin_amdgcn_wmma_f32_16x16x32_bf16(
                  false, a.v, false, b.v, (short)0, acc, false, false);
    }

    // Branchless per-lane argmax over the 8 owned logits.
    // Index grows with r, so strict '>' keeps the smallest index on ties.
    float best = acc[0];
    int   bidx = mbase;
    #pragma unroll
    for (int r = 1; r < 8; ++r) {
        const float v = acc[r];
        if (v > best) { best = v; bidx = mbase + r; }
    }
    #pragma unroll
    for (int off = 16; off; off >>= 1) {
        const float ov = __shfl_down(best, off, 32);
        const int   oi = __shfl_down(bidx, off, 32);
        if (ov > best || (ov == best && oi < bidx)) { best = ov; bidx = oi; }
    }
    if (lane == 0) {
        // value in high bits; (V - idx) in low bits so ties prefer smaller idx
        const unsigned long long k =
            ((unsigned long long)fkey(best) << 32) | (unsigned)(V - bidx);
        atomicMax(key, k);
    }
}

// read argmax winner, emit token, gather next embedding into x
__global__ void finalize_kernel(const unsigned long long* __restrict__ key,
                                const float* __restrict__ emb,
                                float* __restrict__ x,
                                float* __restrict__ out,
                                int step) {
    const unsigned long long k = *key;
    const int idx = V - (int)(unsigned)(k & 0xFFFFFFFFull);
    const int tid = blockIdx.x * blockDim.x + threadIdx.x;
    if (tid < E) x[tid] = emb[(size_t)idx * E + tid];
    if (tid == 0) out[step] = (float)idx;
}

// ---------------- launcher ----------------
extern "C" void kernel_launch(void* const* d_in, const int* in_sizes, int n_in,
                              void* d_out, int out_size, void* d_ws, size_t ws_size,
                              hipStream_t stream) {
    (void)in_sizes; (void)n_in; (void)out_size; (void)ws_size;
    const float* inp   = (const float*)d_in[0];
    const float* W_ih  = (const float*)d_in[1];
    const float* W_hh  = (const float*)d_in[2];
    const float* b_ih  = (const float*)d_in[3];
    const float* b_hh  = (const float*)d_in[4];
    const float* emb   = (const float*)d_in[5];
    const float* W_lin = (const float*)d_in[6];
    const float* b_lin = (const float*)d_in[7];
    float* out = (float*)d_out;

    // workspace layout (all offsets >= 32B aligned)
    char* ws = (char*)d_ws;
    unsigned long long* key = (unsigned long long*)(ws + 0);
    float*          x     = (float*)(ws + 64);                       // E fp32
    float*          h     = (float*)(ws + 64 + E * 4);               // H fp32
    float*          c     = (float*)(ws + 64 + E * 4 + H * 4);       // H fp32
    float*          gates = (float*)(ws + 64 + E * 4 + 2 * H * 4);   // 4H fp32
    unsigned short* hbf   = (unsigned short*)(ws + 64 + E * 4 + 6 * H * 4);  // H bf16
    float*          bpad  = (float*)(ws + 28928);                    // VPAD fp32 bias
    unsigned short* wbf   = (unsigned short*)(ws + 230144);          // VPAD*H bf16 (~98MB)

    // one-time per launch: bf16-ize W_lin + build padded bias, init state
    convert_wlin_kernel<<<(VPAD * H) / (256 * 4), 256, 0, stream>>>(W_lin, wbf);
    bias_pad_kernel<<<(VPAD + 255) / 256, 256, 0, stream>>>(b_lin, bpad);
    init_state_kernel<<<4, 256, 0, stream>>>(inp, x, h, c);

    const int logit_blocks = (NTILES * 32 + 255) / 256;   // 393
    for (int step = 0; step < NUM_STEPS; ++step) {
        gates_kernel<<<4 * H / 8, 256, 0, stream>>>(W_ih, W_hh, b_ih, b_hh, x, h, gates);
        cell_kernel<<<4, 256, 0, stream>>>(gates, c, h, hbf, key);
        logits_argmax_kernel<<<logit_blocks, 256, 0, stream>>>(wbf, hbf, bpad, key);
        finalize_kernel<<<2, 256, 0, stream>>>(key, emb, x, out, step);
    }
}
